// SpaMamba_1417339207785
// MI455X (gfx1250) — compile-verified, matrix operand order checked by reading, not run on hardware
//
#include <hip/hip_runtime.h>
#include <hip/hip_bf16.h>

// ---------------- problem constants ----------------
#define LSEQ    16384            // 128*128 tokens (B=1)
#define DMODEL  128
#define DINNER  256
#define DSTATE  16
#define DTRANK  8
#define DCONV   4
#define NPADXP  48               // x_proj N (=40) padded to 3 WMMA tiles
#define CH      128              // scan chunk length
#define NCHUNK  (LSEQ / CH)      // 128 chunks

typedef __attribute__((ext_vector_type(16))) _Float16 v16h;
typedef __attribute__((ext_vector_type(8)))  _Float16 v8h;
typedef __attribute__((ext_vector_type(8)))  float    v8f;
typedef __attribute__((ext_vector_type(4)))  unsigned int v4u;
typedef __attribute__((ext_vector_type(8)))  int      v8i;
typedef __attribute__((ext_vector_type(4)))  int      v4i;

// ---- Tensor Data Mover availability (device pass only; host pass -> fallback) ----
#if defined(__has_builtin)
# if __has_builtin(__builtin_amdgcn_tensor_load_to_lds) && __has_builtin(__builtin_amdgcn_s_wait_tensorcnt)
#  define TDM_OK 1
# endif
#endif
#ifndef TDM_OK
# define TDM_OK 0
#endif

#if TDM_OK
// Build a 2D tensor-DMA descriptor (D#) per CDNA5 ISA ch.8 and issue the load.
//   lds_off    : byte offset into LDS where tile starts
//   gaddr      : global address of tile start
//   k_elems    : row length (elements of 2 bytes) = tensor_dim0 = tile_dim0
//   rows_rem   : rows remaining in tensor from tile start (OOB rows -> zero fill)
//   tile_rows  : tile_dim1
//   pad_code   : pad_interval code (row dwords = 2^(code+1)); pad_amount fixed 4 dwords (16B)
__device__ __forceinline__ void tdm_load_2d(unsigned lds_off, const void* gaddr,
                                            unsigned k_elems, unsigned rows_rem,
                                            unsigned tile_rows, unsigned pad_code) {
    unsigned long long ga = (unsigned long long)(size_t)gaddr;
    v4u g0;
    g0[0] = 1u;                                        // count=1, user descriptor
    g0[1] = lds_off;                                   // lds_addr
    g0[2] = (unsigned)(ga & 0xffffffffu);              // global_addr[31:0]
    g0[3] = (unsigned)((ga >> 32) & 0x01ffffffu)       // global_addr[56:32]
          | (2u << 30);                                // type = 2 ("image")
    v8i g1;
    g1[0] = (int)((1u << 16)                           // data_size = 1 -> 2 bytes
          | (1u << 20)                                 // pad_enable (LDS bank rotation)
          | (pad_code << 22)                           // pad_interval
          | (3u << 25));                               // pad_amount = 4 dwords (16B)
    g1[1] = (int)((k_elems & 0xffffu) << 16);          // tensor_dim0[15:0]
    g1[2] = (int)((k_elems >> 16) & 0xffffu)           // tensor_dim0[31:16]
          | (int)((rows_rem & 0xffffu) << 16);         // tensor_dim1[15:0]
    g1[3] = (int)((rows_rem >> 16) & 0xffffu)          // tensor_dim1[31:16]
          | (int)((k_elems & 0xffffu) << 16);          // tile_dim0
    g1[4] = (int)(tile_rows & 0xffffu);                // tile_dim1 (tile_dim2 = 0)
    g1[5] = (int)k_elems;                              // tensor_dim0_stride[31:0]
    g1[6] = 0;                                         // stride0 hi / stride1 lo
    g1[7] = 0;                                         // stride1 hi
    v4i z4 = {0, 0, 0, 0};
#if __clang_major__ >= 23
    v8i z8 = {0, 0, 0, 0, 0, 0, 0, 0};
    __builtin_amdgcn_tensor_load_to_lds(g0, g1, z4, z4, z8, 0);
#else
    __builtin_amdgcn_tensor_load_to_lds(g0, g1, z4, z4, 0);
#endif
}
#endif // TDM_OK

// ---------------- weight conversion fp32 -> fp16 ----------------
__global__ void k_cvt_inproj(const float* __restrict__ w, _Float16* __restrict__ wh) {
    int i = blockIdx.x * 256 + threadIdx.x;           // 512*128
    if (i < 512 * DMODEL) wh[i] = (_Float16)w[i];
}
__global__ void k_cvt_xproj(const float* __restrict__ w, _Float16* __restrict__ wh) {
    int i = blockIdx.x * 256 + threadIdx.x;           // padded [48][256]
    if (i < NPADXP * DINNER) {
        int n = i >> 8;
        wh[i] = (n < DTRANK + 2 * DSTATE) ? (_Float16)w[i] : (_Float16)0.f;
    }
}
__global__ void k_cvt_outproj(const float* __restrict__ w, _Float16* __restrict__ wh) {
    int i = blockIdx.x * 256 + threadIdx.x;           // 128*256
    if (i < DMODEL * DINNER) wh[i] = (_Float16)w[i];
}

// ---------------- CHW -> (L, C) transpose + f16 convert (LDS tiled) ----------------
__global__ __launch_bounds__(256) void k_transpose_x(const float* __restrict__ x,
                                                     _Float16* __restrict__ xfh) {
    __shared__ float tile[16][17];
    const int lb = blockIdx.x << 4;
    const int cb = blockIdx.y << 4;
    const int tx = threadIdx.x & 15;
    const int ty = threadIdx.x >> 4;
    tile[ty][tx] = x[(size_t)(cb + ty) * LSEQ + lb + tx];
    __syncthreads();
    xfh[(size_t)(lb + ty) * DMODEL + cb + tx] = (_Float16)tile[tx][ty];
}

// ---------------- WMMA fragment gather from a pitched row ----------------
__device__ __forceinline__ v16h ld_frag(const _Float16* p) {
    v8h lo = *(const v8h*)p;
    v8h hi = *(const v8h*)(p + 16);
    v16h r;
#pragma unroll
    for (int i = 0; i < 8; ++i) { r[i] = lo[i]; r[8 + i] = hi[i]; }
    return r;
}

// ---------------- TDM-staged WMMA GEMM: C[M,Nld] = A[M,K] * W[N,K]^T ----------------
// Block tile 128(M) x 32(N): 8 waves, wave w = m-tile w, covering both n-tiles
// (A fragment reused across two WMMAs). A/B panels staged into LDS by the
// Tensor Data Mover with pad_enable rotating banks (pitch = K+8 halves).
// GUARD=false for tile-aligned N (straight-line clause-able store epilogue).
template<int K, bool GUARD>
__global__ __launch_bounds__(256) void k_gemm_wmma(const _Float16* __restrict__ A,
                                                   const _Float16* __restrict__ W,
                                                   float* __restrict__ C,
                                                   int N, int Nld) {
    constexpr int PITCH = K + 8;                      // +16B per row via TDM padding
    extern __shared__ _Float16 smem[];
    _Float16* As = smem;                              // [128][PITCH]
    _Float16* Bs = smem + 128 * PITCH;                // [32][PITCH]

    const int lane = threadIdx.x & 31;
    const int wave = threadIdx.x >> 5;
    const int mb = blockIdx.y * 128;
    const int nb = blockIdx.x * 32;

#if TDM_OK
    if (threadIdx.x == 0) {
        const unsigned padc = (K == 128) ? 5u : 6u;   // row dwords: 64 / 128
        tdm_load_2d((unsigned)(size_t)As, A + (size_t)mb * K, K,
                    (unsigned)(LSEQ - mb), 128u, padc);
        tdm_load_2d((unsigned)(size_t)Bs, W + (size_t)nb * K, K,
                    (unsigned)(N > nb ? N - nb : 0), 32u, padc);
        __builtin_amdgcn_s_wait_tensorcnt(0);
    }
#else
    for (int idx = threadIdx.x; idx < 128 * (K / 8); idx += 256) {
        int r = idx / (K / 8), cc = (idx % (K / 8)) * 8;
        *(v8h*)&As[r * PITCH + cc] = *(const v8h*)&A[(size_t)(mb + r) * K + cc];
    }
    for (int idx = threadIdx.x; idx < 32 * (K / 8); idx += 256) {
        int r = idx / (K / 8), cc = (idx % (K / 8)) * 8;
        v8h v = {};
        if (nb + r < N) v = *(const v8h*)&W[(size_t)(nb + r) * K + cc];
        *(v8h*)&Bs[r * PITCH + cc] = v;
    }
#endif
    __syncthreads();

    const int ml = (wave << 4) + (lane & 15);         // A row within block tile
    const int nl = lane & 15;                         // B row within n-tile
    const int kb = (lane >> 4) << 3;                  // 0 or 8: fragment k-offset
    v8f acc0 = {}, acc1 = {};
    for (int k0 = 0; k0 < K; k0 += 32) {
        v16h a  = ld_frag(&As[ml * PITCH + k0 + kb]);
        v16h b0 = ld_frag(&Bs[nl * PITCH + k0 + kb]);
        v16h b1 = ld_frag(&Bs[(16 + nl) * PITCH + k0 + kb]);
        acc0 = __builtin_amdgcn_wmma_f32_16x16x32_f16(false, a, false, b0,
                                                      (short)0, acc0, false, false);
        acc1 = __builtin_amdgcn_wmma_f32_16x16x32_f16(false, a, false, b1,
                                                      (short)0, acc1, false, false);
    }

    // C/D layout: lanes 0-15 rows 0..7, lanes 16-31 rows 8..15; col = lane&15
    const int rbase = mb + (wave << 4) + ((lane >> 4) << 3);
    const int col0 = nb + nl, col1 = nb + 16 + nl;
    float* __restrict__ cp = C + (size_t)rbase * Nld + col0;
    if (GUARD) {
        if (col0 < N) {
#pragma unroll
            for (int r = 0; r < 8; ++r) cp[(size_t)r * Nld] = acc0[r];
        }
        if (col1 < N) {
#pragma unroll
            for (int r = 0; r < 8; ++r) cp[(size_t)r * Nld + 16] = acc1[r];
        }
    } else {
#pragma unroll
        for (int r = 0; r < 8; ++r) {
            cp[(size_t)r * Nld]      = acc0[r];
            cp[(size_t)r * Nld + 16] = acc1[r];
        }
    }
}

// ---------------- depthwise causal conv1d (k=4) + bias + SiLU ----------------
__global__ void k_conv_silu(const float* __restrict__ xz, const float* __restrict__ cw,
                            const float* __restrict__ cb, float* __restrict__ xm,
                            _Float16* __restrict__ xmh) {
    int i = blockIdx.x * 256 + threadIdx.x;           // L * DINNER
    if (i >= LSEQ * DINNER) return;
    int l = i >> 8, c = i & 255;
    float acc = cb[c];
#pragma unroll
    for (int j = 0; j < DCONV; ++j) {
        int ll = l - (DCONV - 1) + j;
        if (ll >= 0) acc += xz[(size_t)ll * 512 + c] * cw[c * DCONV + j];
    }
    float s = acc * (1.f / (1.f + __expf(-acc)));     // SiLU
    xm[i] = s;
    xmh[i] = (_Float16)s;
}

// ---------------- dt = softplus(dtr @ dt_proj^T + b), K = 8 (VALU) ----------------
__global__ void k_dt(const float* __restrict__ dbl, const float* __restrict__ dtw,
                     const float* __restrict__ dtb, float* __restrict__ dt) {
    int i = blockIdx.x * 256 + threadIdx.x;           // L * DINNER
    if (i >= LSEQ * DINNER) return;
    int l = i >> 8, c = i & 255;
    float s = dtb[c];
#pragma unroll
    for (int r = 0; r < DTRANK; ++r)
        s += dbl[(size_t)l * NPADXP + r] * dtw[c * DTRANK + r];
    dt[i] = (s > 20.f) ? s : log1pf(__expf(s));
}

// ---------------- selective scan, pass A: per-chunk local scan ----------------
__global__ void k_scan_local(const float* __restrict__ dt, const float* __restrict__ xm,
                             const float* __restrict__ dbl, const float* __restrict__ A_log,
                             float* __restrict__ Q, float* __restrict__ Sdt) {
    int t = blockIdx.x * 256 + threadIdx.x;           // NCHUNK * DINNER
    if (t >= NCHUNK * DINNER) return;
    int c = t & 255, ck = t >> 8;
    float Ac[DSTATE], h[DSTATE];
#pragma unroll
    for (int s = 0; s < DSTATE; ++s) { Ac[s] = -__expf(A_log[c * DSTATE + s]); h[s] = 0.f; }
    float sdt = 0.f;
    for (int i = 0; i < CH; ++i) {
        int l = ck * CH + i;
        float d  = dt[(size_t)l * DINNER + c];
        float xv = xm[(size_t)l * DINNER + c];
        sdt += d;
        float dx = d * xv;
#pragma unroll
        for (int s = 0; s < DSTATE; ++s) {
            float bm = dbl[(size_t)l * NPADXP + DTRANK + s];
            h[s] = __expf(d * Ac[s]) * h[s] + dx * bm;
        }
    }
#pragma unroll
    for (int s = 0; s < DSTATE; ++s) Q[(size_t)ck * 4096 + c * DSTATE + s] = h[s];
    Sdt[ck * DINNER + c] = sdt;
}

// ---------------- pass B: chunk prefix combine (decay = exp(A * sum(dt))) ----------------
__global__ void k_scan_prefix(const float* __restrict__ A_log, const float* __restrict__ Sdt,
                              const float* __restrict__ Q, float* __restrict__ Hinit) {
    int t = blockIdx.x * 256 + threadIdx.x;           // DINNER * DSTATE = 4096
    if (t >= DINNER * DSTATE) return;
    int c = t >> 4, s = t & 15;
    float Ac = -__expf(A_log[c * DSTATE + s]);
    float hp = 0.f;
    for (int ck = 0; ck < NCHUNK; ++ck) {
        Hinit[(size_t)ck * 4096 + c * DSTATE + s] = hp;
        hp = __expf(Ac * Sdt[ck * DINNER + c]) * hp + Q[(size_t)ck * 4096 + c * DSTATE + s];
    }
}

// ---------------- pass C: replay + C-proj + D skip + SiLU gate -> f16 ----------------
__global__ void k_scan_final(const float* __restrict__ dt, const float* __restrict__ xm,
                             const float* __restrict__ dbl, const float* __restrict__ A_log,
                             const float* __restrict__ Dp, const float* __restrict__ xz,
                             const float* __restrict__ Hinit, _Float16* __restrict__ yh) {
    int t = blockIdx.x * 256 + threadIdx.x;           // NCHUNK * DINNER
    if (t >= NCHUNK * DINNER) return;
    int c = t & 255, ck = t >> 8;
    float Ac[DSTATE], h[DSTATE];
#pragma unroll
    for (int s = 0; s < DSTATE; ++s) {
        Ac[s] = -__expf(A_log[c * DSTATE + s]);
        h[s]  = Hinit[(size_t)ck * 4096 + c * DSTATE + s];
    }
    const float dcoef = Dp[c];
    for (int i = 0; i < CH; ++i) {
        int l = ck * CH + i;
        float d  = dt[(size_t)l * DINNER + c];
        float xv = xm[(size_t)l * DINNER + c];
        float dx = d * xv;
        float y = 0.f;
#pragma unroll
        for (int s = 0; s < DSTATE; ++s) {
            float bm = dbl[(size_t)l * NPADXP + DTRANK + s];
            float cm = dbl[(size_t)l * NPADXP + DTRANK + DSTATE + s];
            h[s] = __expf(d * Ac[s]) * h[s] + dx * bm;
            y += h[s] * cm;
        }
        y += dcoef * xv;
        float z = xz[(size_t)l * 512 + DINNER + c];
        y *= z * (1.f / (1.f + __expf(-z)));          // * silu(z)
        yh[(size_t)l * DINNER + c] = (_Float16)y;
    }
}

// ---------------- BatchNorm stats: one block per channel ----------------
__global__ __launch_bounds__(256) void k_bn_stats(const float* __restrict__ ob,
                                                  float* __restrict__ stats) {
    __shared__ float ss[256], sq[256];
    const int c = blockIdx.x, tid = threadIdx.x;
    float s = 0.f, q = 0.f;
    for (int l = tid; l < LSEQ; l += 256) {
        float v = ob[(size_t)l * DMODEL + c];
        s += v; q += v * v;
    }
    ss[tid] = s; sq[tid] = q; __syncthreads();
    for (int o = 128; o > 0; o >>= 1) {
        if (tid < o) { ss[tid] += ss[tid + o]; sq[tid] += sq[tid + o]; }
        __syncthreads();
    }
    if (tid == 0) {
        float mu  = ss[0] / (float)LSEQ;
        float var = sq[0] / (float)LSEQ - mu * mu;
        stats[c * 2]     = mu;
        stats[c * 2 + 1] = rsqrtf(var + 1e-5f);
    }
}

// ---------------- BN affine + LeakyReLU + residual, (L,C) -> (C,H,W) ----------------
__global__ void k_bn_lrelu_res(const float* __restrict__ ob, const float* __restrict__ stats,
                               const float* __restrict__ g, const float* __restrict__ b,
                               const float* __restrict__ x, float* __restrict__ out) {
    int i = blockIdx.x * 256 + threadIdx.x;           // L * DMODEL
    if (i >= LSEQ * DMODEL) return;
    int l = i >> 7, c = i & 127;
    float v = (ob[i] - stats[c * 2]) * stats[c * 2 + 1] * g[c] + b[c];
    v = (v > 0.f) ? v : 0.2f * v;
    out[(size_t)c * LSEQ + l] = v + x[(size_t)c * LSEQ + l];
}

// ---------------- host orchestration ----------------
extern "C" void kernel_launch(void* const* d_in, const int* in_sizes, int n_in,
                              void* d_out, int out_size, void* d_ws, size_t ws_size,
                              hipStream_t stream) {
    const float* x         = (const float*)d_in[0];
    const float* in_proj_w = (const float*)d_in[1];
    const float* conv_w    = (const float*)d_in[2];
    const float* conv_b    = (const float*)d_in[3];
    const float* x_proj_w  = (const float*)d_in[4];
    const float* dt_proj_w = (const float*)d_in[5];
    const float* dt_proj_b = (const float*)d_in[6];
    const float* A_log     = (const float*)d_in[7];
    const float* Dp        = (const float*)d_in[8];
    const float* out_proj_w= (const float*)d_in[9];
    const float* bn_g      = (const float*)d_in[10];
    const float* bn_b      = (const float*)d_in[11];
    float* out = (float*)d_out;

    char* ws = (char*)d_ws;
    _Float16* wh_in  = (_Float16*)(ws + 0);                       // 512x128
    _Float16* wh_xp  = (_Float16*)(ws + (131072));                // 48x256 padded
    _Float16* wh_out = (_Float16*)(ws + (131072 + 24576));        // 128x256
    size_t off = 262144;
    _Float16* xf_h  = (_Float16*)(ws + off); off += (size_t)LSEQ * DMODEL * 2;  // 4 MB
    float*    xz    = (float*)   (ws + off); off += (size_t)LSEQ * 512 * 4;     // 32 MB
    float*    xm    = (float*)   (ws + off); off += (size_t)LSEQ * DINNER * 4;  // 16 MB
    _Float16* xm_h  = (_Float16*)(ws + off); off += (size_t)LSEQ * DINNER * 2;  // 8 MB
    float*    dbl   = (float*)   (ws + off); off += (size_t)LSEQ * NPADXP * 4;  // 3 MB
    float*    dt    = (float*)   (ws + off); off += (size_t)LSEQ * DINNER * 4;  // 16 MB
    float*    Q     = (float*)   (ws + off); off += (size_t)NCHUNK * 4096 * 4;  // 2 MB
    float*    Sdt   = (float*)   (ws + off); off += (size_t)NCHUNK * DINNER * 4;
    float*    Hinit = (float*)   (ws + off); off += (size_t)NCHUNK * 4096 * 4;  // 2 MB
    _Float16* yh    = (_Float16*)(ws + off); off += (size_t)LSEQ * DINNER * 2;  // 8 MB
    float*    outb  = (float*)   (ws + off); off += (size_t)LSEQ * DMODEL * 4;  // 8 MB
    float*    stats = (float*)   (ws + off);

    // weight + activation conversions
    k_cvt_inproj <<<256, 256, 0, stream>>>(in_proj_w, wh_in);
    k_cvt_xproj  <<<48,  256, 0, stream>>>(x_proj_w, wh_xp);
    k_cvt_outproj<<<128, 256, 0, stream>>>(out_proj_w, wh_out);
    k_transpose_x<<<dim3(LSEQ / 16, DMODEL / 16), 256, 0, stream>>>(x, xf_h);

    // dynamic LDS: (128 + 32) rows * (K + 8) halves * 2 bytes
    const size_t lds128 = (size_t)160 * (128 + 8) * 2;   // 43,520 B
    const size_t lds256 = (size_t)160 * (256 + 8) * 2;   // 84,480 B

    // in_proj: [L,128] x [512,128]^T -> [L,512]
    k_gemm_wmma<DMODEL, false><<<dim3(512 / 32, LSEQ / 128), 256, lds128, stream>>>(
        xf_h, wh_in, xz, 512, 512);

    // causal depthwise conv + SiLU
    k_conv_silu<<<(LSEQ * DINNER) / 256, 256, 0, stream>>>(xz, conv_w, conv_b, xm, xm_h);

    // x_proj: [L,256] x [40,256]^T -> [L,48] (N=40 live, 2 blocks cover 4 n-tiles)
    k_gemm_wmma<DINNER, true><<<dim3(2, LSEQ / 128), 256, lds256, stream>>>(
        xm_h, wh_xp, dbl, DTRANK + 2 * DSTATE, NPADXP);

    // dt projection + softplus
    k_dt<<<(LSEQ * DINNER) / 256, 256, 0, stream>>>(dbl, dt_proj_w, dt_proj_b, dt);

    // chunked selective scan
    k_scan_local <<<(NCHUNK * DINNER) / 256, 256, 0, stream>>>(dt, xm, dbl, A_log, Q, Sdt);
    k_scan_prefix<<<(DINNER * DSTATE) / 256, 256, 0, stream>>>(A_log, Sdt, Q, Hinit);
    k_scan_final <<<(NCHUNK * DINNER) / 256, 256, 0, stream>>>(dt, xm, dbl, A_log, Dp, xz,
                                                               Hinit, yh);

    // out_proj: [L,256] x [128,256]^T -> [L,128]
    k_gemm_wmma<DINNER, false><<<dim3(DMODEL / 32, LSEQ / 128), 256, lds256, stream>>>(
        yh, wh_out, outb, DMODEL, DMODEL);

    // BatchNorm (batch stats) + LeakyReLU + residual
    k_bn_stats    <<<DMODEL, 256, 0, stream>>>(outb, stats);
    k_bn_lrelu_res<<<(LSEQ * DMODEL) / 256, 256, 0, stream>>>(outb, stats, bn_g, bn_b, x, out);
}
